// ProductQuantizer_74698071212066
// MI455X (gfx1250) — matrix operand order repaired and checked by reading, not compiled, not run on hardware
//
#include <hip/hip_runtime.h>
#include <hip/hip_bf16.h>

typedef __attribute__((ext_vector_type(16))) _Float16 v16h;
typedef __attribute__((ext_vector_type(8)))  _Float16 v8h;
typedef __attribute__((ext_vector_type(8)))  float    v8f;

#define B_   8
#define L_   2048
#define H_   8
#define M_   1024
#define D_   64
#define N_   (B_ * L_)          // 16384 rows per head
#define ROWS 128                // rows (l values) handled per block
#define RMS_EPS 1.1920929e-07f
#define TOTAL_ELEMS 8388608.0f  // H*N*d, denominator of the mean

// ---------------------------------------------------------------------------
// Zero workspace accumulators (sums, counts, sse) -- graph-capture safe.
// ---------------------------------------------------------------------------
__global__ void vq_zero(float* __restrict__ sums, float* __restrict__ counts,
                        float* __restrict__ sse, int nsum, int ncnt) {
    int i = blockIdx.x * blockDim.x + threadIdx.x;
    if (i < nsum) sums[i] = 0.0f;
    if (i < ncnt) counts[i] = 0.0f;
    if (i == 0)   sse[0] = 0.0f;
}

// ---------------------------------------------------------------------------
// Convert codebooks f32 -> f16 (row-major [H][M][64]) for WMMA A operands.
// ---------------------------------------------------------------------------
__global__ void vq_cvt(const float* __restrict__ cb, _Float16* __restrict__ cb16, int n) {
    int i = blockIdx.x * blockDim.x + threadIdx.x;
    if (i < n) cb16[i] = (_Float16)cb[i];
}

// ---------------------------------------------------------------------------
// Main fused kernel: async-LDS z staging + sim GEMM (WMMA f16) + argmax +
// z_q scatter + loss SSE + EMA segment sums.
// Block = (head h, 128 consecutive l for one b). 256 threads = 8 waves.
// A = codebook chunk (16 codes x 64), B = z rows (64 x 16) so C holds, per
// lane, 8 candidate codes for ONE fixed z-row -> argmax is per-lane.
// ---------------------------------------------------------------------------
__global__ __launch_bounds__(256)
void vq_main(const float* __restrict__ z, const float* __restrict__ cb,
             const _Float16* __restrict__ cb16,
             float* __restrict__ zq_out, float* __restrict__ idx_out,
             float* __restrict__ sums, float* __restrict__ counts,
             float* __restrict__ sse) {
    __shared__ __align__(16) float zt32[D_ * ROWS];  // [dd][l] f32, 32 KB
    __shared__ int   sidx[ROWS];
    __shared__ float red[8];

    const int bx   = blockIdx.x;        // 0..1023
    const int h    = bx >> 7;           // 128 blocks per head
    const int tile = bx & 127;
    const int b    = tile >> 4;         // 16 tiles of 128 rows per batch (16*128 = L)
    const int l0   = (tile & 15) << 7;
    const int t    = threadIdx.x;

    const float* zbase = z + ((size_t)(b * 512 + h * 64) * L_) + l0;

    // ---- Stage z tile into LDS with the CDNA5 async DMA path -------------
    // Tile = 64 dd-rows x 128 l (f32, matches global layout: contiguous in l).
    // 2048 x 16B transfers / 256 threads = 8 async b128 loads per thread.
    #pragma unroll
    for (int i = 0; i < 8; ++i) {
        int c  = i * 256 + t;           // 16-byte chunk id
        int dd = c >> 5;                // /32 chunks per dd-row
        int lc = (c & 31) << 2;         // starting l of this chunk
        const float* gptr = zbase + (size_t)dd * L_ + lc;
        unsigned lds = (unsigned)(size_t)&zt32[dd * ROWS + lc];  // LDS byte offset
        asm volatile("global_load_async_to_lds_b128 %0, %1, off"
                     :: "v"(lds), "v"(gptr) : "memory");
    }
    asm volatile("s_wait_asynccnt 0x0" ::: "memory");
    __syncthreads();

    const int wave  = t >> 5;
    const int lane  = t & 31;
    const int col   = lane & 15;        // z-row within the wave's 16-row tile
    const int hi    = lane >> 4;        // lane group selects K/code half
    const int myrow = wave * 16 + col;  // row (l) within block tile [0,128)

    // ---- Build B operands (z rows as f16), fixed for the whole M loop ----
    // B layout: lanes 0-15 hold K=kb+0..15, lanes 16-31 K=kb+16..31.
    const int kb = hi * 16;
    v16h bop0, bop1;
    #pragma unroll
    for (int j = 0; j < 16; ++j) {
        bop0[j] = (_Float16)zt32[(kb + j) * ROWS + myrow];       // K 0..31
        bop1[j] = (_Float16)zt32[(32 + kb + j) * ROWS + myrow];  // K 32..63
    }

    const _Float16* cbh = cb16 + (size_t)h * M_ * D_;
    const int ka      = hi * 16;  // A lane K-base: lanes<16 -> {0..15}, else {16..31}
    const int codeOff = hi * 8;  // C VGPR v -> code = chunk*16 + v (+8 for hi lanes)

    // Per-accumulator-slot running max + chunk id: no index math in the loop
    // (chunk is wave-uniform -> SGPR cndmask source).
    float bestv[8];
    int   bchk[8];
    #pragma unroll
    for (int v = 0; v < 8; ++v) { bestv[v] = -3.4e38f; bchk[v] = 0; }

    for (int chunk = 0; chunk < 64; ++chunk) {
        const _Float16* arow = cbh + (size_t)(chunk * 16 + col) * D_;
        if (chunk + 1 < 64)
            __builtin_prefetch(arow + 16 * D_, 0, 1);  // global_prefetch_b8 next chunk

        v8h a00 = *(const v8h*)(arow + ka);            // K ka..ka+7
        v8h a01 = *(const v8h*)(arow + ka + 8);        // K ka+8..ka+15
        v8h a10 = *(const v8h*)(arow + 32 + ka);
        v8h a11 = *(const v8h*)(arow + 32 + ka + 8);
        v16h a0, a1;
        #pragma unroll
        for (int j = 0; j < 8; ++j) {
            a0[j] = a00[j]; a0[j + 8] = a01[j];
            a1[j] = a10[j]; a1[j + 8] = a11[j];
        }

        v8f c = {};
        c = __builtin_amdgcn_wmma_f32_16x16x32_f16(false, a0, false, bop0,
                                                   (short)0, c, false, false);
        c = __builtin_amdgcn_wmma_f32_16x16x32_f16(false, a1, false, bop1,
                                                   (short)0, c, false, false);

        #pragma unroll
        for (int v = 0; v < 8; ++v) {
            float s = c[v];
            if (s > bestv[v]) { bestv[v] = s; bchk[v] = chunk; }
        }
    }

    // Combine the 8 slots (tie -> lower code = first occurrence).
    float best = -3.4e38f;
    int   bidx = 0x7fffffff;
    #pragma unroll
    for (int v = 0; v < 8; ++v) {
        int code = bchk[v] * 16 + codeOff + v;
        float s  = bestv[v];
        if (s > best || (s == best && code < bidx)) { best = s; bidx = code; }
    }

    // Lanes l and l^16 hold the same z-row with disjoint code halves: merge.
    float obest = __shfl_xor(best, 16, 32);
    int   oidx  = __shfl_xor(bidx, 16, 32);
    if (obest > best || (obest == best && oidx < bidx)) { best = obest; bidx = oidx; }

    if (lane < 16) sidx[myrow] = bidx;
    __syncthreads();

    // Indices out: layout [B][H][L] (written as float values).
    if (t < ROWS)
        idx_out[((size_t)(b * H_ + h)) * L_ + l0 + t] = (float)sidx[t];

    // Cooperative z_q scatter (coalesced along l), loss SSE, EMA segment
    // sums. z is re-read from LDS (no second HBM pass).
    const float* cbf = cb + (size_t)h * M_ * D_;
    float* zqbase = zq_out + ((size_t)(b * 512 + h * 64) * L_) + l0;
    float lsum = 0.0f;
    #pragma unroll 4
    for (int i = 0; i < 32; ++i) {
        int e  = i * 256 + t;
        int dd = e >> 7;
        int l  = e & 127;
        int m  = sidx[l];
        float q  = cbf[(size_t)m * D_ + dd];
        float zv = zt32[dd * ROWS + l];
        zqbase[(size_t)dd * L_ + l] = q;  // straight-through value == quantized
        float diff = zv - q;
        lsum += diff * diff;
        atomicAdd(&sums[((size_t)h * M_ + m) * D_ + dd], zv);
    }
    if (t < ROWS) atomicAdd(&counts[h * M_ + sidx[t]], 1.0f);

    // Block-reduce SSE: wave32 shuffle tree, then one atomic per block.
    #pragma unroll
    for (int off = 16; off > 0; off >>= 1) lsum += __shfl_xor(lsum, off, 32);
    if (lane == 0) red[wave] = lsum;
    __syncthreads();
    if (t == 0) {
        float s = 0.0f;
        #pragma unroll
        for (int w = 0; w < 8; ++w) s += red[w];
        atomicAdd(sse, s);
    }
}

// ---------------------------------------------------------------------------
// Codebook EMA update: one block per (h,m), 64 threads (one per d element).
// mean -> slerp(mean, codebook, 0.99) -> rms_norm, gated by count>0.
// ---------------------------------------------------------------------------
__global__ __launch_bounds__(64)
void vq_update(const float* __restrict__ cb, const float* __restrict__ sums,
               const float* __restrict__ counts, float* __restrict__ out_cb) {
    __shared__ float rr[3][2];
    const int hm   = blockIdx.x;     // 0..8191
    const int t    = threadIdx.x;    // 0..63
    const int wave = t >> 5, lane = t & 31;

    float cnt  = counts[hm];
    float high = cb[(size_t)hm * D_ + t];
    float low  = sums[(size_t)hm * D_ + t] / fmaxf(cnt, 1.0f);

    float pd = low * high, pl = low * low, ph = high * high;
    #pragma unroll
    for (int off = 16; off > 0; off >>= 1) {
        pd += __shfl_xor(pd, off, 32);
        pl += __shfl_xor(pl, off, 32);
        ph += __shfl_xor(ph, off, 32);
    }
    if (lane == 0) { rr[0][wave] = pd; rr[1][wave] = pl; rr[2][wave] = ph; }
    __syncthreads();
    float dot = rr[0][0] + rr[0][1];
    float nl  = sqrtf(rr[1][0] + rr[1][1]);
    float nh  = sqrtf(rr[2][0] + rr[2][1]);

    float cosv  = dot / fmaxf(nl * nh, 1e-8f);
    cosv        = fminf(fmaxf(cosv, -1.0f + 1e-7f), 1.0f - 1e-7f);
    float omega = acosf(cosv);
    float so    = sinf(omega);
    float x = (low * sinf(0.01f * omega) + high * sinf(0.99f * omega)) / so;

    float px = x * x;
    #pragma unroll
    for (int off = 16; off > 0; off >>= 1) px += __shfl_xor(px, off, 32);
    __syncthreads();
    if (lane == 0) rr[0][wave] = px;
    __syncthreads();
    float ms = (rr[0][0] + rr[0][1]) * (1.0f / 64.0f);
    float nx = x / sqrtf(ms + RMS_EPS);

    out_cb[(size_t)hm * D_ + t] = (cnt > 0.0f) ? nx : high;
}

// vq_loss = codebook_loss + 0.25*commitment_loss; both equal the same MSE.
__global__ void vq_finalize(const float* __restrict__ sse, float* __restrict__ loss) {
    if (threadIdx.x == 0 && blockIdx.x == 0)
        loss[0] = sse[0] * 1.25f / TOTAL_ELEMS;
}

extern "C" void kernel_launch(void* const* d_in, const int* in_sizes, int n_in,
                              void* d_out, int out_size, void* d_ws, size_t ws_size,
                              hipStream_t stream) {
    const float* z  = (const float*)d_in[0];   // [8,512,2048] f32
    const float* cb = (const float*)d_in[1];   // [8,1024,64]  f32

    // Workspace layout (~3.2 MB)
    char* ws = (char*)d_ws;
    _Float16* cb16 = (_Float16*)ws;                                  // 1 MB
    float* sums    = (float*)(ws + (1u << 20));                      // 2 MB  [H][M][64]
    float* counts  = (float*)(ws + (1u << 20) + (2u << 20));         // 32 KB [H][M]
    float* sse     = (float*)(ws + (1u << 20) + (2u << 20) + (32u << 10));

    // d_out: z_q | vq_loss | indices | new_codebooks (flat, return order)
    float* zq   = (float*)d_out;           // 8388608
    float* loss = zq + 8388608;            // 1
    float* idxo = loss + 1;                // 131072  [B][H][L]
    float* ncb  = idxo + 131072;           // 524288  [H][M][64]

    const int ncb_elems = H_ * M_ * D_;    // 524288
    vq_zero<<<(ncb_elems + 255) / 256, 256, 0, stream>>>(sums, counts, sse,
                                                         ncb_elems, H_ * M_);
    vq_cvt<<<(ncb_elems + 255) / 256, 256, 0, stream>>>(cb, cb16, ncb_elems);
    vq_main<<<H_ * (N_ / ROWS), 256, 0, stream>>>(z, cb, cb16, zq, idxo,
                                                  sums, counts, sse);
    vq_update<<<H_ * M_, 64, 0, stream>>>(cb, sums, counts, ncb);
    vq_finalize<<<1, 64, 0, stream>>>(sse, loss);
}
